// ResNetWithHARichness_64244120813925
// MI455X (gfx1250) — matrix operand order repaired and checked by reading, not compiled, not run on hardware
//
#include <hip/hip_runtime.h>
#include <math.h>

// ---------------------------------------------------------------------------
// MI455X (gfx1250) ResNet-50 + hue-attention + richness forward.
// NHWC f16 activations, f32 WMMA accum (v_wmma_f32_16x16x32_f16), BN folded,
// BN+ReLU+residual fused in the GEMM epilogue.
// Weights pre-packed into a contiguous-k A-fragment layout (k = 16*khalf + e); the workgroup
// stages each 32x32 A K-tile into LDS with GLOBAL_LOAD_ASYNC_TO_LDS_B64
// (double-buffered, ASYNCcnt pipelined) and all 8 waves share it.
// Per wave: 32(M) x 64(N) tile -> 8 WMMAs per 32-deep K step, B reused 2x.
// ---------------------------------------------------------------------------

typedef __attribute__((ext_vector_type(16))) _Float16 v16h;
typedef __attribute__((ext_vector_type(8)))  float    v8f;

__device__ inline v16h zero16h() {
    v16h z;
#pragma unroll
    for (int i = 0; i < 16; ++i) z[i] = (_Float16)0.0f;
    return z;
}
__device__ inline v8f zero8f() {
    v8f z;
#pragma unroll
    for (int i = 0; i < 8; ++i) z[i] = 0.0f;
    return z;
}

// --------------------------- utility kernels -------------------------------

__global__ __launch_bounds__(256) void k_zero(float* __restrict__ p, int n) {
    int i = blockIdx.x * 256 + threadIdx.x;
    if (i < n) p[i] = 0.0f;
}

// Fold BN: scale = gamma*rsqrt(var+eps); bias = beta - mean*scale
__global__ __launch_bounds__(256) void k_fold(const float* __restrict__ g,
                                              const float* __restrict__ be,
                                              const float* __restrict__ me,
                                              const float* __restrict__ va,
                                              float* __restrict__ sc,
                                              float* __restrict__ bi, int C) {
    int i = blockIdx.x * 256 + threadIdx.x;
    if (i < C) {
        float s = g[i] * rsqrtf(va[i] + 1e-5f);
        sc[i] = s;
        bi[i] = be[i] - me[i] * s;
    }
}

// Pack OIHW fp32 weights -> f16 A-fragment order.
// Fragment element (kt, o, khalf, e) maps to K = kt*32 + khalf*16 + e
// with K decomposed as k = (ky*KW + kx)*Cin + c   (NHWC implicit-GEMM K order).
__global__ __launch_bounds__(256) void k_pack(const float* __restrict__ w,
                                              _Float16* __restrict__ dst,
                                              int O, int Cin, int KH, int KW) {
    long long K = (long long)KH * KW * Cin;
    long long total = (long long)O * K;
    long long d = (long long)blockIdx.x * 256 + threadIdx.x;
    if (d >= total) return;
    int e = (int)(d & 15);
    long long t = d >> 4;
    int khalf = (int)(t & 1); t >>= 1;
    int o = (int)(t % O);
    int kt = (int)(t / O);
    int k = kt * 32 + khalf * 16 + e;
    int c = k % Cin;
    int r = k / Cin;
    int ky = r / KW;
    int kx = r - ky * KW;
    dst[d] = (_Float16)w[(((long long)o * Cin + c) * KH + ky) * KW + kx];
}

// conv1: 7x7 s2 p3, Cin=3 (too thin for WMMA K; direct VALU, tiny fraction of work)
__global__ __launch_bounds__(256) void k_conv1(const float* __restrict__ x,
                                               const float* __restrict__ w,
                                               const float* __restrict__ sc,
                                               const float* __restrict__ bi,
                                               _Float16* __restrict__ out, int B) {
    int idx = blockIdx.x * 256 + threadIdx.x;
    int total = B * 112 * 112 * 64;
    if (idx >= total) return;
    int o = idx & 63;
    int t = idx >> 6;
    int ox = t % 112; t /= 112;
    int oy = t % 112;
    int b = t / 112;
    float sum = 0.0f;
#pragma unroll
    for (int c = 0; c < 3; ++c) {
        const float* xp = x + ((long long)(b * 3 + c)) * (224 * 224);
        const float* wp = w + (o * 3 + c) * 49;
        for (int ky = 0; ky < 7; ++ky) {
            int iy = oy * 2 - 3 + ky;
            if ((unsigned)iy >= 224u) continue;
            for (int kx = 0; kx < 7; ++kx) {
                int ix = ox * 2 - 3 + kx;
                if ((unsigned)ix >= 224u) continue;
                sum += xp[iy * 224 + ix] * wp[ky * 7 + kx];
            }
        }
    }
    float val = sum * sc[o] + bi[o];
    out[idx] = (_Float16)(val > 0.0f ? val : 0.0f);
}

// maxpool 3x3 s2 p1 on NHWC f16
__global__ __launch_bounds__(256) void k_maxpool(const _Float16* __restrict__ in,
                                                 _Float16* __restrict__ out,
                                                 int B, int C, int Hin, int Win,
                                                 int Ho, int Wo) {
    int idx = blockIdx.x * 256 + threadIdx.x;
    int total = B * Ho * Wo * C;
    if (idx >= total) return;
    int c = idx % C;
    int t = idx / C;
    int ox = t % Wo; t /= Wo;
    int oy = t % Ho;
    int b = t / Ho;
    float m = -3.0e38f;
    for (int dy = 0; dy < 3; ++dy) {
        int iy = oy * 2 - 1 + dy;
        if ((unsigned)iy >= (unsigned)Hin) continue;
        for (int dx = 0; dx < 3; ++dx) {
            int ix = ox * 2 - 1 + dx;
            if ((unsigned)ix >= (unsigned)Win) continue;
            float v = (float)in[(((long long)b * Hin + iy) * Win + ix) * C + c];
            m = v > m ? v : m;
        }
    }
    out[idx] = (_Float16)m;
}

// ------------------------- WMMA implicit-GEMM conv -------------------------
// Block = 8 waves; block tile = 32(M) x 512(N). The 32x32 f16 A K-tile (2 KB)
// is async-copied to LDS (double-buffered) and shared by all waves.
// Per wave per K-step: 2 LDS A fragments + 4 global B fragments -> 8 WMMAs.
__global__ __launch_bounds__(256) void k_conv_wmma(
    const _Float16* __restrict__ act, const _Float16* __restrict__ wp,
    const float* __restrict__ sc, const float* __restrict__ bi,
    const _Float16* __restrict__ res, _Float16* __restrict__ out,
    int B, int Hin, int Win, int Cin, int Ho, int Wo, int O,
    int KH, int KW, int stride, int pad, int relu) {
    __shared__ _Float16 smA[2][1024];  // 2 x (32 M x 32 K) f16

    const int Ntot = B * Ho * Wo;
    const int mPairs = O >> 5;                 // O is a multiple of 32
    const int nGroups = (Ntot + 63) >> 6;
    const int tid = threadIdx.x;
    const int waveId = tid >> 5;
    const int lane = tid & 31;

    const int mp = blockIdx.x % mPairs;
    const int ns = blockIdx.x / mPairs;
    int ng = ns * 8 + waveId;
    if (ng >= nGroups) ng = nGroups - 1;       // clamp: duplicate waves rewrite
                                               // identical values; keeps barrier
                                               // participation + EXEC uniform
    const int mBase = mp << 5;
    const int nBase = ng << 6;
    const int lm = lane & 15;
    const int khalf = lane >> 4;

    int bImg[4], iy0[4], ix0[4], nIdx[4];
#pragma unroll
    for (int s = 0; s < 4; ++s) {
        int n = nBase + (s << 4) + lm;
        nIdx[s] = n;
        int nc = n < Ntot ? n : Ntot - 1;
        int bI = nc / (Ho * Wo);
        int rem = nc - bI * (Ho * Wo);
        int oy = rem / Wo;
        int ox = rem - oy * Wo;
        bImg[s] = bI;
        iy0[s] = oy * stride - pad;
        ix0[s] = ox * stride - pad;
    }

    v8f acc[8];
#pragma unroll
    for (int s = 0; s < 8; ++s) acc[s] = zero8f();
    const v16h vz = zero16h();

    const int K = KH * KW * Cin;
    const int kTiles = K >> 5;

    // Async A-tile staging: 256 threads x 8B = 2KB per K-tile.
    const unsigned long long wpBase = (unsigned long long)(size_t)wp;
    const unsigned lds0 = (unsigned)(size_t)&smA[0][0];
    const unsigned lds1 = (unsigned)(size_t)&smA[1][0];
    const unsigned mByte = (unsigned)(mBase * 32) * 2u;  // byte offset of mPair slab

    auto stage = [&](int kt, int buf) {
        unsigned gOff = (unsigned)((unsigned long long)kt * O * 64ull) + mByte + tid * 8u;
        unsigned lAddr = (buf ? lds1 : lds0) + tid * 8u;
        // GLOBAL_LOAD_ASYNC_TO_LDS_B64 (GVS mode): vdst = LDS byte addr,
        // vaddr = u32 offset, saddr = 64-bit base. Tracked by ASYNCcnt.
        asm volatile("global_load_async_to_lds_b64 %0, %1, %2"
                     :: "v"(lAddr), "v"(gOff), "s"(wpBase)
                     : "memory");
    };

    stage(0, 0);
    for (int kt = 0; kt < kTiles; ++kt) {
        const int cur = kt & 1;
        asm volatile("s_wait_asynccnt 0" ::: "memory");  // own wave's copy done
        __syncthreads();                                 // everyone's copy done
        if (kt + 1 < kTiles) stage(kt + 1, 1 - cur);     // overlap next copy

        // A fragments from LDS (layout mirrors packed global layout)
        const _Float16* sa = &smA[cur][0];
        v16h a0 = *(const v16h*)(sa + ((lm << 1) + khalf) * 16);
        v16h a1 = *(const v16h*)(sa + (((16 + lm) << 1) + khalf) * 16);

        int k0 = (kt << 5) + (khalf << 4);
        int rr = k0 / Cin;  // 16-chunk never crosses (ky,kx): Cin % 32 == 0
        int c0 = k0 - rr * Cin;
        int ky = rr / KW;
        int kx = rr - ky * KW;
#pragma unroll
        for (int s = 0; s < 4; ++s) {
            int iy = iy0[s] + ky;
            int ix = ix0[s] + kx;
            v16h bf;
            if ((unsigned)iy < (unsigned)Hin && (unsigned)ix < (unsigned)Win) {
                bf = *(const v16h*)(act +
                     (((long long)bImg[s] * Hin + iy) * Win + ix) * Cin + c0);
            } else {
                bf = vz;  // zero padding
            }
            acc[s] = __builtin_amdgcn_wmma_f32_16x16x32_f16(
                false, a0, false, bf, (short)0, acc[s], false, false);
            acc[4 + s] = __builtin_amdgcn_wmma_f32_16x16x32_f16(
                false, a1, false, bf, (short)0, acc[4 + s], false, false);
        }
        __syncthreads();  // all waves done reading smA[cur] before its reuse
    }

    // Epilogue: D row m = v + 8*khalf, col n = lane&15 (ISA 16x16 f32 C/D layout)
    float s16[16], b16[16];
#pragma unroll
    for (int ms = 0; ms < 2; ++ms) {
#pragma unroll
        for (int v = 0; v < 8; ++v) {
            int o = mBase + (ms << 4) + (khalf << 3) + v;
            s16[ms * 8 + v] = sc[o];
            b16[ms * 8 + v] = bi[o];
        }
    }
#pragma unroll
    for (int s = 0; s < 4; ++s) {
        int n = nIdx[s];
        if (n < Ntot) {
#pragma unroll
            for (int ms = 0; ms < 2; ++ms) {
                long long rowOff = (long long)n * O + mBase + (ms << 4) + (khalf << 3);
#pragma unroll
                for (int v = 0; v < 8; ++v) {
                    float val = acc[ms * 4 + s][v] * s16[ms * 8 + v] + b16[ms * 8 + v];
                    if (res) val += (float)res[rowOff + v];
                    if (relu) val = val > 0.0f ? val : 0.0f;
                    out[rowOff + v] = (_Float16)val;
                }
            }
        }
    }
}

// global average pool: NHWC f16 (B,H,W,C) -> feats[b*2050 + c] (fp32)
__global__ __launch_bounds__(256) void k_avgpool(const _Float16* __restrict__ in,
                                                 float* __restrict__ feats,
                                                 int B, int C, int H, int W) {
    int idx = blockIdx.x * 256 + threadIdx.x;
    if (idx >= B * C) return;
    int c = idx % C;
    int b = idx / C;
    float sum = 0.0f;
    for (int y = 0; y < H; ++y)
        for (int x = 0; x < W; ++x)
            sum += (float)in[(((long long)b * H + y) * W + x) * C + c];
    feats[(long long)b * 2050 + c] = sum / (float)(H * W);
}

// hue branch: per-pixel hue -> 1x1 conv relu -> 1x1 conv sigmoid -> h*a, sum
__global__ __launch_bounds__(256) void k_hue(const float* __restrict__ x,
                                             const float* __restrict__ w1,
                                             const float* __restrict__ b1,
                                             const float* __restrict__ w2,
                                             const float* __restrict__ b2,
                                             float* __restrict__ ghAcc) {
    __shared__ float sd[256];
    const int HW = 224 * 224;
    int b = blockIdx.y;
    int i = blockIdx.x * 256 + threadIdx.x;
    float contrib = 0.0f;
    if (i < HW) {
        float r = x[((long long)b * 3 + 0) * HW + i];
        float g = x[((long long)b * 3 + 1) * HW + i];
        float bl = x[((long long)b * 3 + 2) * HW + i];
        float maxc = fmaxf(r, fmaxf(g, bl));
        float minc = fminf(r, fminf(g, bl));
        float delta = maxc - minc;
        float safe = (delta == 0.0f) ? 1.0f : delta;
        float h;
        if (maxc == r)      h = 60.0f * (g - bl) / safe;
        else if (maxc == g) h = 120.0f + 60.0f * (bl - r) / safe;
        else                h = 240.0f + 60.0f * (r - g) / safe;
        if (delta == 0.0f) h = 0.0f;
        if (h < 0.0f) h += 360.0f;
        float a = h * w1[0] + b1[0];
        a = a > 0.0f ? a : 0.0f;
        a = a * w2[0] + b2[0];
        a = 1.0f / (1.0f + expf(-a));
        contrib = h * a;
    }
    sd[threadIdx.x] = contrib;
    __syncthreads();
    for (int s = 128; s > 0; s >>= 1) {
        if (threadIdx.x < s) sd[threadIdx.x] += sd[threadIdx.x + s];
        __syncthreads();
    }
    if (threadIdx.x == 0) atomicAdd(&ghAcc[b], sd[0]);
}

// richness: 256-bin LDS histogram per (b,c), count nonzero bins
__global__ __launch_bounds__(256) void k_rich(const float* __restrict__ x,
                                              float* __restrict__ richAcc) {
    __shared__ unsigned int hist[256];
    __shared__ float sd[256];
    const int HW = 224 * 224;
    int c = blockIdx.x;
    int b = blockIdx.y;
    hist[threadIdx.x] = 0u;
    __syncthreads();
    const float* xp = x + ((long long)b * 3 + c) * HW;
    for (int i = threadIdx.x; i < HW; i += 256) {
        float v = xp[i] * 255.0f;
        int bin = (int)floorf(v * (256.0f / 255.0f));
        bin = bin < 0 ? 0 : (bin > 255 ? 255 : bin);
        atomicAdd(&hist[bin], 1u);
    }
    __syncthreads();
    sd[threadIdx.x] = hist[threadIdx.x] > 0u ? 1.0f : 0.0f;
    __syncthreads();
    for (int s = 128; s > 0; s >>= 1) {
        if (threadIdx.x < s) sd[threadIdx.x] += sd[threadIdx.x + s];
        __syncthreads();
    }
    if (threadIdx.x == 0) atomicAdd(&richAcc[b], sd[0]);
}

__global__ __launch_bounds__(64) void k_finalize(const float* __restrict__ ghAcc,
                                                 const float* __restrict__ richAcc,
                                                 float* __restrict__ feats, int B) {
    int b = threadIdx.x;
    if (b >= B) return;
    feats[(long long)b * 2050 + 2048] = ghAcc[b] / (224.0f * 224.0f) * 3.0f;  // ALPHA
    feats[(long long)b * 2050 + 2049] = (richAcc[b] / 3.0f) * 2.0f;           // BETA
}

// classifier: out[b][k] = feats[b] . W[k] + bias[k]
__global__ __launch_bounds__(256) void k_fc(const float* __restrict__ feats,
                                            const float* __restrict__ w,
                                            const float* __restrict__ bvec,
                                            float* __restrict__ out) {
    __shared__ float sd[256];
    int b = blockIdx.x >> 1;
    int k = blockIdx.x & 1;
    float sum = 0.0f;
    for (int j = threadIdx.x; j < 2050; j += 256)
        sum += feats[(long long)b * 2050 + j] * w[k * 2050 + j];
    sd[threadIdx.x] = sum;
    __syncthreads();
    for (int s = 128; s > 0; s >>= 1) {
        if (threadIdx.x < s) sd[threadIdx.x] += sd[threadIdx.x + s];
        __syncthreads();
    }
    if (threadIdx.x == 0) out[b * 2 + k] = sd[0] + bvec[k];
}

// --------------------------------- host ------------------------------------

extern "C" void kernel_launch(void* const* d_in, const int* in_sizes, int n_in,
                              void* d_out, int out_size, void* d_ws, size_t ws_size,
                              hipStream_t stream) {
    (void)in_sizes; (void)n_in; (void)out_size; (void)ws_size;
    int idx = 0;
    auto nxt = [&]() { return (const float*)d_in[idx++]; };

    // ---- walk inputs in setup_inputs() insertion order ----
    const float* X = nxt();                                  // (32,3,224,224)
    const float* c1w = nxt();                                // conv1 (64,3,7,7)
    const float* c1g = nxt(); const float* c1b = nxt();
    const float* c1m = nxt(); const float* c1v = nxt();      // bn1

    struct CD { const float *w, *g, *be, *me, *va; int Cin, O, KH, KW;
                long long packOff, scOff; };
    CD convs[64]; int nconv = 0;
    struct BD { int c1, c2, c3, cd, stride; };
    BD blocks[16]; int nb = 0;

    const int stage_w[4] = {64, 128, 256, 512};
    const int stage_n[4] = {3, 4, 6, 3};
    int in_c = 64;
    for (int si = 0; si < 4; ++si) {
        int width = stage_w[si], out_c = width * 4;
        for (int bi2 = 0; bi2 < stage_n[si]; ++bi2) {
            BD bd; bd.stride = (si > 0 && bi2 == 0) ? 2 : 1;
            CD c;
            c.w = nxt(); c.g = nxt(); c.be = nxt(); c.me = nxt(); c.va = nxt();
            c.Cin = in_c; c.O = width; c.KH = 1; c.KW = 1;
            bd.c1 = nconv; convs[nconv++] = c;
            c.w = nxt(); c.g = nxt(); c.be = nxt(); c.me = nxt(); c.va = nxt();
            c.Cin = width; c.O = width; c.KH = 3; c.KW = 3;
            bd.c2 = nconv; convs[nconv++] = c;
            c.w = nxt(); c.g = nxt(); c.be = nxt(); c.me = nxt(); c.va = nxt();
            c.Cin = width; c.O = out_c; c.KH = 1; c.KW = 1;
            bd.c3 = nconv; convs[nconv++] = c;
            bd.cd = -1;
            if (bi2 == 0) {
                c.w = nxt(); c.g = nxt(); c.be = nxt(); c.me = nxt(); c.va = nxt();
                c.Cin = in_c; c.O = out_c; c.KH = 1; c.KW = 1;
                bd.cd = nconv; convs[nconv++] = c;
            }
            blocks[nb++] = bd;
            in_c = out_c;
        }
    }
    const float* hw1 = nxt(); const float* hb1 = nxt();
    const float* hw2 = nxt(); const float* hb2 = nxt();      // hue
    const float* clsW = nxt(); const float* clsB = nxt();    // cls

    const int B = 32;

    // ---- workspace bump allocator ----
    char* ws = (char*)d_ws;
    size_t off = 0;
    auto alloc = [&](size_t bytes) {
        size_t a = (off + 255) & ~(size_t)255;
        off = a + bytes;
        return a;
    };
    const size_t ACT_ELEMS = (size_t)B * 112 * 112 * 64;  // largest map
    size_t bOff[3];
    for (int i = 0; i < 3; ++i) bOff[i] = alloc(ACT_ELEMS * 2);

    long long packElems = 0, scElems = 0;
    long long c1sc = scElems; scElems += 64 * 2;
    for (int i = 0; i < nconv; ++i) {
        convs[i].packOff = packElems;
        packElems += (long long)convs[i].O * convs[i].KH * convs[i].KW * convs[i].Cin;
        convs[i].scOff = scElems;
        scElems += (long long)convs[i].O * 2;
    }
    size_t packOffB = alloc((size_t)packElems * 2);
    size_t scOffB   = alloc((size_t)scElems * 4);
    size_t featOff  = alloc((size_t)B * 2050 * 4);
    size_t accOff   = alloc((size_t)B * 2 * 4);

    _Float16* bufs[3] = { (_Float16*)(ws + bOff[0]), (_Float16*)(ws + bOff[1]),
                          (_Float16*)(ws + bOff[2]) };
    _Float16* packBase = (_Float16*)(ws + packOffB);
    float* scBase = (float*)(ws + scOffB);
    float* feats  = (float*)(ws + featOff);
    float* ghAcc  = (float*)(ws + accOff);
    float* richAcc = ghAcc + B;

    // ---- prep: zero accumulators, fold BN, pack weights ----
    k_zero<<<1, 256, 0, stream>>>(ghAcc, 2 * B);
    float* c1scale = scBase + c1sc;
    float* c1bias  = c1scale + 64;
    k_fold<<<1, 256, 0, stream>>>(c1g, c1b, c1m, c1v, c1scale, c1bias, 64);
    for (int i = 0; i < nconv; ++i) {
        CD& c = convs[i];
        long long total = (long long)c.O * c.KH * c.KW * c.Cin;
        k_pack<<<(int)((total + 255) / 256), 256, 0, stream>>>(
            c.w, packBase + c.packOff, c.O, c.Cin, c.KH, c.KW);
        k_fold<<<(c.O + 255) / 256, 256, 0, stream>>>(
            c.g, c.be, c.me, c.va, scBase + c.scOff, scBase + c.scOff + c.O, c.O);
    }

    // ---- stem ----
    {
        long long total = (long long)B * 112 * 112 * 64;
        k_conv1<<<(int)((total + 255) / 256), 256, 0, stream>>>(
            X, c1w, c1scale, c1bias, bufs[0], B);
        long long tp = (long long)B * 56 * 56 * 64;
        k_maxpool<<<(int)((tp + 255) / 256), 256, 0, stream>>>(
            bufs[0], bufs[1], B, 64, 112, 112, 56, 56);
    }

    _Float16 *bi_ = bufs[1], *bj_ = bufs[0], *bk_ = bufs[2];
    int H = 56, W = 56;

    auto runConv = [&](int ci, const _Float16* src, _Float16* dst,
                       const _Float16* resPtr, int Hin, int Win, int Ho, int Wo,
                       int stride, int pad, int relu) {
        CD& c = convs[ci];
        int Ntot = B * Ho * Wo;
        int mPairs = c.O >> 5;
        int nGroups = (Ntot + 63) >> 6;
        int nSuper = (nGroups + 7) / 8;
        int grid = mPairs * nSuper;
        k_conv_wmma<<<grid, 256, 0, stream>>>(
            src, packBase + c.packOff, scBase + c.scOff, scBase + c.scOff + c.O,
            resPtr, dst, B, Hin, Win, c.Cin, Ho, Wo, c.O, c.KH, c.KW,
            stride, pad, relu);
    };

    // ---- bottleneck blocks (3-buffer rotation keeps residual live) ----
    for (int bx = 0; bx < nb; ++bx) {
        BD& bd = blocks[bx];
        int s = bd.stride;
        int Ho = H / s, Wo = W / s;
        if (bd.cd >= 0) {
            runConv(bd.cd, bi_, bk_, nullptr, H, W, Ho, Wo, s, 0, 0);  // shortcut
            runConv(bd.c1, bi_, bj_, nullptr, H, W, H, W, 1, 0, 1);
            runConv(bd.c2, bj_, bi_, nullptr, H, W, Ho, Wo, s, 1, 1);
            runConv(bd.c3, bi_, bj_, bk_, Ho, Wo, Ho, Wo, 1, 0, 1);    // +res +relu
            _Float16* nin = bj_; _Float16* f1 = bi_; _Float16* f2 = bk_;
            bi_ = nin; bj_ = f1; bk_ = f2;
        } else {
            runConv(bd.c1, bi_, bj_, nullptr, H, W, H, W, 1, 0, 1);
            runConv(bd.c2, bj_, bk_, nullptr, H, W, H, W, 1, 1, 1);
            runConv(bd.c3, bk_, bj_, bi_, H, W, H, W, 1, 0, 1);        // identity res
            _Float16* nin = bj_; _Float16* f1 = bi_; _Float16* f2 = bk_;
            bi_ = nin; bj_ = f1; bk_ = f2;
        }
        H = Ho; W = Wo;
    }

    // ---- head ----
    k_avgpool<<<(B * 2048 + 255) / 256, 256, 0, stream>>>(bi_, feats, B, 2048, H, W);
    {
        dim3 g((224 * 224 + 255) / 256, B);
        k_hue<<<g, 256, 0, stream>>>(X, hw1, hb1, hw2, hb2, ghAcc);
    }
    {
        dim3 g(3, B);
        k_rich<<<g, 256, 0, stream>>>(X, richAcc);
    }
    k_finalize<<<1, 64, 0, stream>>>(ghAcc, richAcc, feats, B);
    k_fc<<<B * 2, 256, 0, stream>>>(feats, clsW, clsB, (float*)d_out);
}
